// Model_SINE_61469571940788
// MI455X (gfx1250) — matrix operand order repaired and checked by reading, not compiled, not run on hardware
//
#include <hip/hip_runtime.h>
#include <hip/hip_bf16.h>
#include <math.h>

// SINE model for MI455X (gfx1250, wave32, WMMA).
// B=1024 T=256 D=128 K=4 CAND=500. Heavy GEMMs run as bf16 WMMA (f32 accum).

#define B_    1024
#define T_    256
#define D_    128
#define K_    4
#define CAND_ 500
#define NEGV  (-4294967295.0f)
#define EPS_  1e-3f

typedef __bf16 bf16;
typedef __attribute__((ext_vector_type(16))) __bf16 v16bf;
typedef __attribute__((ext_vector_type(8)))  float  v8f;

union V16 { v16bf v; uint4 q[2]; };

// ---------------------------------------------------------------------------
// Core WMMA tile: H[16x128] = A[16x128] (bf16, row stride 128, 16B aligned)
//                           @ Wt[128x128]^T-stored (bf16, row n contiguous in k)
// Block = 128 threads (4 waves). Wave w computes column tiles n0 = 32w, 32w+16.
// Per-lane operand packing follows CDNA5 ISA 7.12.2 (16-bit A 16x32 / B 32x16).
// ---------------------------------------------------------------------------
__device__ __forceinline__ void wmma_row_tile(const bf16* __restrict__ Arows,
                                              const bf16* __restrict__ Wt,
                                              float* __restrict__ Hlds) {
  const int lane = threadIdx.x & 31;
  const int wave = threadIdx.x >> 5;
  const int rw   = lane & 15;   // A row / B column within tile
  const int kh   = lane >> 4;   // K-half select
#pragma unroll
  for (int nt = 0; nt < 2; ++nt) {
    const int n0 = wave * 32 + nt * 16;
    v8f acc = {0.f, 0.f, 0.f, 0.f, 0.f, 0.f, 0.f, 0.f};
#pragma unroll
    for (int kk = 0; kk < 4; ++kk) {
      V16 A, Bv;
      // A: lane holds row rw; elems 0..7 -> k = 32kk+8kh+{0..7}; 8..15 -> +16
      const bf16* ap = Arows + rw * D_ + kk * 32 + kh * 8;
      A.q[0] = *(const uint4*)(ap);
      A.q[1] = *(const uint4*)(ap + 16);
      // B: lane holds column n0+rw; elems 0..7 -> k = 32kk+8kh+{0..7}; 8..15 -> +16
      const bf16* bp = Wt + (n0 + rw) * D_ + kk * 32 + kh * 8;
      Bv.q[0] = *(const uint4*)(bp);
      Bv.q[1] = *(const uint4*)(bp + 16);
      acc = __builtin_amdgcn_wmma_f32_16x16x32_bf16(
          false, A.v, false, Bv.v, (short)0, acc, false, false);
    }
#pragma unroll
    for (int r = 0; r < 8; ++r)                 // D: VGPR r -> row r + 8*kh
      Hlds[(r + 8 * kh) * D_ + n0 + rw] = acc[r];
  }
}

// ---------------------------------------------------------------------------
// Stage 0: x = item_his + pos  ->  bf16 staging (64MB, L2-resident)
// ---------------------------------------------------------------------------
__global__ __launch_bounds__(256) void k_prep_x(const float* __restrict__ his,
                                                const float* __restrict__ pos,
                                                bf16* __restrict__ xb) {
  long i = (long)blockIdx.x * 256 + threadIdx.x;
  if (i < (long)B_ * T_ * D_) xb[i] = (bf16)(his[i] + pos[i % (T_ * D_)]);
}

__global__ __launch_bounds__(256) void k_prep_w(const float* W1, const float* W3,
                                                const float* Wk1, bf16* W1T,
                                                bf16* W3T, bf16* Wk1T) {
  int i = blockIdx.x * 256 + threadIdx.x;
  if (i < D_ * D_) {
    int a = i / D_, e = i % D_;
    W1T[a * D_ + e] = (bf16)W1[e * D_ + a];
    W3T[a * D_ + e] = (bf16)W3[e * D_ + a];
  }
  if (i < K_ * D_ * D_) {
    int k = i / (D_ * D_), r = i % (D_ * D_), a = r / D_, e = r % D_;
    Wk1T[k * D_ * D_ + a * D_ + e] = (bf16)Wk1[k * D_ * D_ + e * D_ + a];
  }
}

// ---------------------------------------------------------------------------
// att[b,t] = W2 . tanh(x @ W1)   (WMMA per 16-row tile)
// ---------------------------------------------------------------------------
__global__ __launch_bounds__(128) void k_att(const bf16* __restrict__ xb,
                                             const bf16* __restrict__ W1T,
                                             const float* __restrict__ W2,
                                             float* __restrict__ att) {
  __shared__ float H[16 * D_];
  __shared__ float red[16][8];
  int b = blockIdx.x >> 4, t0 = (blockIdx.x & 15) * 16;
  wmma_row_tile(xb + ((long)b * T_ + t0) * D_, W1T, H);
  __syncthreads();
  int m = threadIdx.x >> 3, sub = threadIdx.x & 7;
  float p = 0.f;
  for (int j = 0; j < 16; ++j) {
    int a = sub * 16 + j;
    p += W2[a] * tanhf(H[m * D_ + a]);
  }
  red[m][sub] = p;
  __syncthreads();
  if (threadIdx.x < 16) {
    float s = 0.f;
    for (int j = 0; j < 8; ++j) s += red[threadIdx.x][j];
    att[(long)b * T_ + t0 + threadIdx.x] = s;
  }
}

// ---------------------------------------------------------------------------
// masked softmax over T, then z_u = sum_t a_t * x_t
// ---------------------------------------------------------------------------
__global__ __launch_bounds__(256) void k_soft_z(const float* __restrict__ att,
                                                const int* __restrict__ mask,
                                                const float* __restrict__ his,
                                                const float* __restrict__ pos,
                                                float* __restrict__ zu) {
  __shared__ float s[256], aS[256], zp[256];
  int b = blockIdx.x, t = threadIdx.x;
  float v = att[(long)b * T_ + t];
  if (mask[(long)b * T_ + t] == 0) v = NEGV;
  s[t] = v; __syncthreads();
  for (int o = 128; o > 0; o >>= 1) { if (t < o) s[t] = fmaxf(s[t], s[t + o]); __syncthreads(); }
  float mx = s[0]; __syncthreads();
  float e = expf(v - mx); s[t] = e; __syncthreads();
  for (int o = 128; o > 0; o >>= 1) { if (t < o) s[t] += s[t + o]; __syncthreads(); }
  aS[t] = e / s[0]; __syncthreads();
  int d = t & 127, h = t >> 7;
  float acc = 0.f;
  for (int j = 0; j < 128; ++j) {
    int tt = h * 128 + j;
    acc += aS[tt] * (his[((long)b * T_ + tt) * D_ + d] + pos[tt * D_ + d]);
  }
  zp[t] = acc; __syncthreads();
  if (t < 128) zu[(long)b * D_ + t] = zp[t] + zp[t + 128];
}

// ---------------------------------------------------------------------------
// s_u = z_u @ C^T, top-4, c_u = C[idx]*sigmoid(s), ln_cu = LN(c_u; g2,b2)
// ---------------------------------------------------------------------------
__global__ __launch_bounds__(256) void k_topk(const float* __restrict__ zu,
                                              const float* __restrict__ C,
                                              float* __restrict__ cu,
                                              float* __restrict__ lncu,
                                              const float* __restrict__ g2,
                                              const float* __restrict__ b2) {
  __shared__ float sv[512];
  __shared__ float rv[256];
  __shared__ int ri[256];
  __shared__ float topv[K_]; __shared__ int topi[K_];
  __shared__ float mu[K_], rs[K_];
  int b = blockIdx.x, t = threadIdx.x;
  const float* z = zu + (long)b * D_;
  for (int c = t; c < 512; c += 256) {
    float acc = -INFINITY;
    if (c < CAND_) {
      acc = 0.f;
      for (int d = 0; d < D_; ++d) acc += z[d] * C[c * D_ + d];
    }
    sv[c] = acc;
  }
  __syncthreads();
  for (int k = 0; k < K_; ++k) {
    float v0 = sv[t], v1 = sv[t + 256];
    if (v1 > v0) { rv[t] = v1; ri[t] = t + 256; } else { rv[t] = v0; ri[t] = t; }
    __syncthreads();
    for (int o = 128; o > 0; o >>= 1) {
      if (t < o && rv[t + o] > rv[t]) { rv[t] = rv[t + o]; ri[t] = ri[t + o]; }
      __syncthreads();
    }
    if (t == 0) { topv[k] = rv[0]; topi[k] = ri[0]; sv[ri[0]] = -INFINITY; }
    __syncthreads();
  }
  for (int e = t; e < K_ * D_; e += 256) {
    int k = e >> 7, d = e & 127;
    float sg = 1.f / (1.f + expf(-topv[k]));
    cu[((long)b * K_ + k) * D_ + d] = C[topi[k] * D_ + d] * sg;
  }
  __syncthreads();
  if (t < K_) {
    const float* p = cu + ((long)b * K_ + t) * D_;
    float s1 = 0.f, s2 = 0.f;
    for (int d = 0; d < D_; ++d) { float x = p[d]; s1 += x; s2 += x * x; }
    float m = s1 / D_; mu[t] = m; rs[t] = rsqrtf(s2 / D_ - m * m + EPS_);
  }
  __syncthreads();
  for (int e = t; e < K_ * D_; e += 256) {
    int k = e >> 7, d = e & 127;
    float x = cu[((long)b * K_ + k) * D_ + d];
    lncu[((long)b * K_ + k) * D_ + d] = (x - mu[k]) * rs[k] * g2[d] + b2[d];
  }
}

// ---------------------------------------------------------------------------
// h1 = LN(x @ W3; g1,b1) . ln_cu  -> masked softmax over K -> P_kt [B,K,T]
// ---------------------------------------------------------------------------
__global__ __launch_bounds__(128) void k_h1(const bf16* __restrict__ xb,
                                            const bf16* __restrict__ W3T,
                                            const float* __restrict__ g1,
                                            const float* __restrict__ b1v,
                                            const float* __restrict__ lncu,
                                            const int* __restrict__ mask,
                                            float* __restrict__ Pkt) {
  __shared__ float H[16 * D_];
  __shared__ float muS[16], rsS[16];
  __shared__ float L[16][K_];
  int b = blockIdx.x >> 4, t0 = (blockIdx.x & 15) * 16;
  wmma_row_tile(xb + ((long)b * T_ + t0) * D_, W3T, H);
  __syncthreads();
  if (threadIdx.x < 16) {
    float s1 = 0.f, s2 = 0.f;
    for (int a = 0; a < D_; ++a) { float x = H[threadIdx.x * D_ + a]; s1 += x; s2 += x * x; }
    float m = s1 / D_; muS[threadIdx.x] = m; rsS[threadIdx.x] = rsqrtf(s2 / D_ - m * m + EPS_);
  }
  __syncthreads();
  if (threadIdx.x < 64) {
    int m = threadIdx.x >> 2, k = threadIdx.x & 3;
    const float* lc = lncu + ((long)b * K_ + k) * D_;
    float acc = 0.f;
    for (int a = 0; a < D_; ++a)
      acc += ((H[m * D_ + a] - muS[m]) * rsS[m] * g1[a] + b1v[a]) * lc[a];
    L[m][k] = acc;
  }
  __syncthreads();
  if (threadIdx.x < 16) {
    int m = threadIdx.x;
    bool pad = (mask[(long)b * T_ + t0 + m] == 0);
    float l0 = pad ? NEGV : L[m][0], l1 = pad ? NEGV : L[m][1];
    float l2 = pad ? NEGV : L[m][2], l3 = pad ? NEGV : L[m][3];
    float mx = fmaxf(fmaxf(l0, l1), fmaxf(l2, l3));
    float e0 = expf(l0 - mx), e1 = expf(l1 - mx), e2 = expf(l2 - mx), e3 = expf(l3 - mx);
    float inv = 1.f / (e0 + e1 + e2 + e3);
    Pkt[((long)b * K_ + 0) * T_ + t0 + m] = e0 * inv;
    Pkt[((long)b * K_ + 1) * T_ + t0 + m] = e1 * inv;
    Pkt[((long)b * K_ + 2) * T_ + t0 + m] = e2 * inv;
    Pkt[((long)b * K_ + 3) * T_ + t0 + m] = e3 * inv;
  }
}

// ---------------------------------------------------------------------------
// h2[b,k,t] = W_k2[k] . tanh(x @ W_k1[k])   (grid.y = k)
// ---------------------------------------------------------------------------
__global__ __launch_bounds__(128) void k_h2(const bf16* __restrict__ xb,
                                            const bf16* __restrict__ Wk1T,
                                            const float* __restrict__ Wk2,
                                            float* __restrict__ h2) {
  __shared__ float H[16 * D_];
  __shared__ float red[16][8];
  int b = blockIdx.x >> 4, t0 = (blockIdx.x & 15) * 16, k = blockIdx.y;
  wmma_row_tile(xb + ((long)b * T_ + t0) * D_, Wk1T + k * D_ * D_, H);
  __syncthreads();
  int m = threadIdx.x >> 3, sub = threadIdx.x & 7;
  float p = 0.f;
  for (int j = 0; j < 16; ++j) {
    int a = sub * 16 + j;
    p += Wk2[k * D_ + a] * tanhf(H[m * D_ + a]);
  }
  red[m][sub] = p;
  __syncthreads();
  if (threadIdx.x < 16) {
    float s = 0.f;
    for (int j = 0; j < 8; ++j) s += red[threadIdx.x][j];
    h2[((long)b * K_ + k) * T_ + t0 + threadIdx.x] = s;
  }
}

// masked softmax over T per (b,k) row, in place -> P_tk
__global__ __launch_bounds__(256) void k_soft_tk(float* __restrict__ h2,
                                                 const int* __restrict__ mask) {
  __shared__ float s[256];
  int row = blockIdx.x, b = row >> 2, t = threadIdx.x;
  float v = h2[(long)row * T_ + t];
  if (mask[(long)b * T_ + t] == 0) v = NEGV;
  s[t] = v; __syncthreads();
  for (int o = 128; o > 0; o >>= 1) { if (t < o) s[t] = fmaxf(s[t], s[t + o]); __syncthreads(); }
  float mx = s[0]; __syncthreads();
  float e = expf(v - mx); s[t] = e; __syncthreads();
  for (int o = 128; o > 0; o >>= 1) { if (t < o) s[t] += s[t + o]; __syncthreads(); }
  h2[(long)row * T_ + t] = e / s[0];
}

// interest_emb[b,k,:] = LN( sum_t Pkt*Ptk * x_t ; g3,b3 )
__global__ __launch_bounds__(128) void k_interest(const float* __restrict__ Pkt,
                                                  const float* __restrict__ Ptk,
                                                  const float* __restrict__ his,
                                                  const float* __restrict__ pos,
                                                  const float* __restrict__ g3,
                                                  const float* __restrict__ b3v,
                                                  float* __restrict__ ie) {
  __shared__ float p[256];
  __shared__ float red[128];
  int row = blockIdx.x, b = row >> 2, d = threadIdx.x;
  for (int t = d; t < T_; t += 128)
    p[t] = Pkt[(long)row * T_ + t] * Ptk[(long)row * T_ + t];
  __syncthreads();
  float acc = 0.f;
  for (int t = 0; t < T_; ++t)
    acc += p[t] * (his[((long)b * T_ + t) * D_ + d] + pos[t * D_ + d]);
  red[d] = acc; __syncthreads();
  for (int o = 64; o > 0; o >>= 1) { if (d < o) red[d] += red[d + o]; __syncthreads(); }
  float mean = red[0] / D_; __syncthreads();
  float c = acc - mean; red[d] = c * c; __syncthreads();
  for (int o = 64; o > 0; o >>= 1) { if (d < o) red[d] += red[d + o]; __syncthreads(); }
  float rstd = rsqrtf(red[0] / D_ + EPS_);
  ie[(long)row * D_ + d] = c * rstd * g3[d] + b3v[d];
}

// ---------------------------------------------------------------------------
// h3[b,t] = W4 . tanh(X_hat @ W3); X_hat tile rebuilt in LDS (rank-4 in K)
// ---------------------------------------------------------------------------
__global__ __launch_bounds__(128) void k_h3(const float* __restrict__ Pkt,
                                            const float* __restrict__ cu,
                                            const bf16* __restrict__ W3T,
                                            const float* __restrict__ W4,
                                            float* __restrict__ h3) {
  __shared__ float cuS[K_][D_];
  __shared__ float pS[16][K_];
  __shared__ bf16 xh[16 * D_];
  __shared__ float H[16 * D_];
  __shared__ float red[16][8];
  int b = blockIdx.x >> 4, t0 = (blockIdx.x & 15) * 16;
  for (int e = threadIdx.x; e < K_ * D_; e += 128)
    cuS[e >> 7][e & 127] = cu[(long)b * K_ * D_ + e];
  if (threadIdx.x < 64) {
    int m = threadIdx.x >> 2, k = threadIdx.x & 3;
    pS[m][k] = Pkt[((long)b * K_ + k) * T_ + t0 + m];
  }
  __syncthreads();
  int d = threadIdx.x;
  for (int m = 0; m < 16; ++m) {
    float v = pS[m][0] * cuS[0][d] + pS[m][1] * cuS[1][d] +
              pS[m][2] * cuS[2][d] + pS[m][3] * cuS[3][d];
    xh[m * D_ + d] = (bf16)v;
  }
  __syncthreads();
  wmma_row_tile(xh, W3T, H);
  __syncthreads();
  int mm = threadIdx.x >> 3, sub = threadIdx.x & 7;
  float p = 0.f;
  for (int j = 0; j < 16; ++j) {
    int a = sub * 16 + j;
    p += W4[a] * tanhf(H[mm * D_ + a]);
  }
  red[mm][sub] = p;
  __syncthreads();
  if (threadIdx.x < 16) {
    float s = 0.f;
    for (int j = 0; j < 8; ++j) s += red[threadIdx.x][j];
    h3[(long)b * T_ + t0 + threadIdx.x] = s;
  }
}

// ---------------------------------------------------------------------------
// final: softmax(h3), c_apt = LN( sum_k (sum_t w_t Pkt[k,t]) * c_u[k] ),
//        e_u = softmax(c_apt.ie/0.1), v_u = sum_k e_u_k * ie_k
// ---------------------------------------------------------------------------
__global__ __launch_bounds__(256) void k_final(const float* __restrict__ h3,
                                               const int* __restrict__ mask,
                                               const float* __restrict__ Pkt,
                                               const float* __restrict__ cu,
                                               const float* __restrict__ ie,
                                               const float* __restrict__ g4,
                                               const float* __restrict__ b4v,
                                               float* __restrict__ out) {
  __shared__ float s[256], w[256];
  __shared__ float sk[K_], ek[K_];
  __shared__ float ieS[K_ * D_];
  __shared__ float ca[D_];
  __shared__ float red[128];
  int b = blockIdx.x, t = threadIdx.x;
  for (int e = t; e < K_ * D_; e += 256) ieS[e] = ie[(long)b * K_ * D_ + e];
  float v = h3[(long)b * T_ + t];
  if (mask[(long)b * T_ + t] == 0) v = NEGV;
  s[t] = v; __syncthreads();
  for (int o = 128; o > 0; o >>= 1) { if (t < o) s[t] = fmaxf(s[t], s[t + o]); __syncthreads(); }
  float mx = s[0]; __syncthreads();
  float ex = expf(v - mx); s[t] = ex; __syncthreads();
  for (int o = 128; o > 0; o >>= 1) { if (t < o) s[t] += s[t + o]; __syncthreads(); }
  w[t] = ex / s[0]; __syncthreads();
  if (t < K_) {
    const float* pk = Pkt + ((long)b * K_ + t) * T_;
    float acc = 0.f;
    for (int tt = 0; tt < T_; ++tt) acc += w[tt] * pk[tt];
    sk[t] = acc;
  }
  __syncthreads();
  if (t < D_) {
    float acc = sk[0] * cu[((long)b * K_ + 0) * D_ + t] +
                sk[1] * cu[((long)b * K_ + 1) * D_ + t] +
                sk[2] * cu[((long)b * K_ + 2) * D_ + t] +
                sk[3] * cu[((long)b * K_ + 3) * D_ + t];
    red[t] = acc; ca[t] = acc;
  }
  __syncthreads();
  for (int o = 64; o > 0; o >>= 1) { if (t < o) red[t] += red[t + o]; __syncthreads(); }
  float mean = red[0] / D_; __syncthreads();
  if (t < D_) { float c = ca[t] - mean; red[t] = c * c; }
  __syncthreads();
  for (int o = 64; o > 0; o >>= 1) { if (t < o) red[t] += red[t + o]; __syncthreads(); }
  float rstd = rsqrtf(red[0] / D_ + EPS_); __syncthreads();
  if (t < D_) ca[t] = (ca[t] - mean) * rstd * g4[t] + b4v[t];
  __syncthreads();
  if (t < K_) {
    float acc = 0.f;
    for (int d = 0; d < D_; ++d) acc += ca[d] * ieS[t * D_ + d];
    ek[t] = acc * 10.f;  // / 0.1
  }
  __syncthreads();
  if (t == 0) {
    float m2 = fmaxf(fmaxf(ek[0], ek[1]), fmaxf(ek[2], ek[3]));
    float e0 = expf(ek[0] - m2), e1 = expf(ek[1] - m2);
    float e2 = expf(ek[2] - m2), e3 = expf(ek[3] - m2);
    float inv = 1.f / (e0 + e1 + e2 + e3);
    ek[0] = e0 * inv; ek[1] = e1 * inv; ek[2] = e2 * inv; ek[3] = e3 * inv;
  }
  __syncthreads();
  if (t < D_)
    out[(long)b * D_ + t] = ek[0] * ieS[0 * D_ + t] + ek[1] * ieS[1 * D_ + t] +
                            ek[2] * ieS[2 * D_ + t] + ek[3] * ieS[3 * D_ + t];
}

// ---------------------------------------------------------------------------
extern "C" void kernel_launch(void* const* d_in, const int* in_sizes, int n_in,
                              void* d_out, int out_size, void* d_ws, size_t ws_size,
                              hipStream_t stream) {
  (void)in_sizes; (void)n_in; (void)out_size; (void)ws_size;
  const float* his  = (const float*)d_in[0];
  const int*   mask = (const int*)d_in[1];
  const float* W1   = (const float*)d_in[2];
  const float* W2   = (const float*)d_in[3];
  const float* W3   = (const float*)d_in[4];
  const float* W4   = (const float*)d_in[5];
  const float* Wk1  = (const float*)d_in[6];
  const float* Wk2  = (const float*)d_in[7];
  const float* C    = (const float*)d_in[8];
  const float* pos  = (const float*)d_in[9];
  const float* g1 = (const float*)d_in[10]; const float* b1 = (const float*)d_in[11];
  const float* g2 = (const float*)d_in[12]; const float* b2 = (const float*)d_in[13];
  const float* g3 = (const float*)d_in[14]; const float* b3 = (const float*)d_in[15];
  const float* g4 = (const float*)d_in[16]; const float* b4 = (const float*)d_in[17];
  float* out = (float*)d_out;

  char* ws = (char*)d_ws;
  size_t off = 0;
  auto alloc = [&](size_t bytes) -> char* {
    char* p = ws + off;
    off += (bytes + 255) & ~size_t(255);
    return p;
  };
  bf16*  xb   = (bf16*)alloc((size_t)B_ * T_ * D_ * 2);      // 64 MB
  bf16*  W1T  = (bf16*)alloc((size_t)D_ * D_ * 2);
  bf16*  W3T  = (bf16*)alloc((size_t)D_ * D_ * 2);
  bf16*  Wk1T = (bf16*)alloc((size_t)K_ * D_ * D_ * 2);
  float* att  = (float*)alloc((size_t)B_ * T_ * 4);
  float* zu   = (float*)alloc((size_t)B_ * D_ * 4);
  float* cu   = (float*)alloc((size_t)B_ * K_ * D_ * 4);
  float* lncu = (float*)alloc((size_t)B_ * K_ * D_ * 4);
  float* Pkt  = (float*)alloc((size_t)B_ * K_ * T_ * 4);
  float* h2   = (float*)alloc((size_t)B_ * K_ * T_ * 4);     // becomes P_tk in place
  float* ie   = (float*)alloc((size_t)B_ * K_ * D_ * 4);
  float* h3   = (float*)alloc((size_t)B_ * T_ * 4);

  k_prep_x <<<(B_ * T_ * D_) / 256, 256, 0, stream>>>(his, pos, xb);
  k_prep_w <<<(K_ * D_ * D_ + 255) / 256, 256, 0, stream>>>(W1, W3, Wk1, W1T, W3T, Wk1T);
  k_att    <<<B_ * 16, 128, 0, stream>>>(xb, W1T, W2, att);
  k_soft_z <<<B_, 256, 0, stream>>>(att, mask, his, pos, zu);
  k_topk   <<<B_, 256, 0, stream>>>(zu, C, cu, lncu, g2, b2);
  k_h1     <<<B_ * 16, 128, 0, stream>>>(xb, W3T, g1, b1, lncu, mask, Pkt);
  dim3 g2d(B_ * 16, K_);
  k_h2     <<<g2d, 128, 0, stream>>>(xb, Wk1T, Wk2, h2);
  k_soft_tk<<<B_ * K_, 256, 0, stream>>>(h2, mask);
  k_interest<<<B_ * K_, 128, 0, stream>>>(Pkt, h2, his, pos, g3, b3, ie);
  k_h3     <<<B_ * 16, 128, 0, stream>>>(Pkt, cu, W3T, W4, h3);
  k_final  <<<B_, 256, 0, stream>>>(h3, mask, Pkt, cu, ie, g4, b4, out);
}
